// POLOAR_Brain_65695819760344
// MI455X (gfx1250) — matrix operand order repaired and checked by reading, not compiled, not run on hardware
//
#include <hip/hip_runtime.h>
#include <hip/hip_bf16.h>
#include <math.h>

// ---------------- problem constants ----------------
#define BB 131072
#define VV 27
#define DD 32
#define HH 64
#define TT 8
#define TAU_INV 1.25f          // 1/0.8
#define LN_EPS 1e-5f
#define LOG2PI 1.8378770664093453f
#define INV_SQRT2 0.70710678118654752f

typedef __attribute__((ext_vector_type(16))) _Float16 v16h;
typedef __attribute__((ext_vector_type(8)))  float    v8f;
typedef __attribute__((ext_vector_type(4)))  float    v4f;

union F16x16 { v16h v; v4f q[2]; };

__device__ __forceinline__ v8f splat8(float x) {
  v8f r;
  #pragma unroll
  for (int i = 0; i < 8; ++i) r[i] = x;
  return r;
}

__device__ __forceinline__ v8f wmma_f16(v16h a, v16h b, v8f c) {
  // D = A(16x32 f16) x B(32x16 f16) + C(16x16 f32)
  return __builtin_amdgcn_wmma_f32_16x16x32_f16(false, a, false, b, (short)0, c,
                                                false, false);
}

// Hardware TRANS32 helpers (v_exp_f32 / v_log_f32 / v_rcp_f32): co-execute
// with VALU and with XDL WMMA ops on CDNA5 (ISA 8.4: trans-tracked).
__device__ __forceinline__ float fexp(float x)  { return __expf(x); }
__device__ __forceinline__ float flog(float x)  { return __logf(x); }
__device__ __forceinline__ float frcp(float x)  { return __builtin_amdgcn_rcpf(x); }
__device__ __forceinline__ float fsigmoid(float x) {
  return frcp(1.0f + fexp(-x));          // exp(-x)->inf => rcp(inf)=0, safe
}
__device__ __forceinline__ float ftanh(float x) {
  const float a = fabsf(x);
  const float e = fexp(-2.0f * a);       // in (0,1], no overflow
  const float r = (1.0f - e) * frcp(1.0f + e);
  return copysignf(r, x);
}

// Load a 16x32 f16 A-fragment from a row-major LDS tile.
// Lane L (L<16): row L, K = 0..7 (elems 0..7) and 16..23 (elems 8..15)
// Lane L>=16:  row L-16, K = 8..15 and 24..31  (ISA 7.12.2 layout)
__device__ __forceinline__ v16h load_a_frag(const _Float16* tile, int rowStride,
                                            int kbase) {
  int lane = threadIdx.x & 31;
  int r    = lane & 15;
  int klo  = (lane >> 4) * 8;
  const _Float16* p = tile + r * rowStride + kbase + klo;
  F16x16 u;
  u.q[0] = *(const v4f*)(p);        // 8 f16, K = klo .. klo+7   (ds_load_b128)
  u.q[1] = *(const v4f*)(p + 16);   // 8 f16, K = klo+16 .. klo+23
  return u.v;
}

// Build a 32x16 f16 B-fragment from a row-major fp32 weight matrix where the
// GEMM is y = x @ W.T, i.e. B[k][n] = W[n][k]. Lane holds column n = ntile*16
// + (lane&15); lanes 0-15 carry K=kbase+0..7,16..23, lanes 16-31 K=kbase+8..15,24..31.
__device__ __forceinline__ v16h load_b_frag(const float* W, int ld, int nmax,
                                            int kmax, int ntile, int kbase) {
  int lane = threadIdx.x & 31;
  int n    = ntile * 16 + (lane & 15);
  int k0   = kbase + ((lane >> 4) * 8);
  F16x16 u;
  #pragma unroll
  for (int e = 0; e < 16; ++e) {
    int k   = k0 + e + ((e >= 8) ? 8 : 0);
    float v = (n < nmax && k < kmax) ? W[n * ld + k] : 0.0f;
    u.v[e]  = (_Float16)v;
  }
  return u.v;
}

// ---------------------------------------------------------------------------
// One wave owns a tile of 16 batch rows. 8 waves / block -> 128 rows / block.
// ---------------------------------------------------------------------------
__global__ __launch_bounds__(256) void POLOAR_Brain_kernel(
    const float* __restrict__ obs, const float* __restrict__ eps,
    const float* __restrict__ u,
    const float* __restrict__ W_enc, const float* __restrict__ b_enc,
    const float* __restrict__ ln_g,  const float* __restrict__ ln_b,
    const float* __restrict__ W_mu,  const float* __restrict__ b_mu,
    const float* __restrict__ W_lv,  const float* __restrict__ b_lv,
    const float* __restrict__ W_ih,  const float* __restrict__ b_ih,
    const float* __restrict__ W_hh,  const float* __restrict__ b_hh,
    const float* __restrict__ W_out, const float* __restrict__ b_out,
    float* __restrict__ out, float* __restrict__ ent_partials) {
  const int lane  = threadIdx.x & 31;
  const int wave  = threadIdx.x >> 5;
  const int c     = lane & 15;          // column within 16-wide N tile
  const int half  = lane >> 4;          // row half for C-fragments
  const int tileB = (blockIdx.x * 8 + wave) * 16;

  __shared__ __align__(16) _Float16 sH[8][16 * HH];   // per-wave staging tile
  __shared__ int   sIdx[8][16];
  __shared__ float sEntW[8];

  _Float16* myH   = &sH[wave][0];
  int*      myIdx = &sIdx[wave][0];

  // ---------------- encoder: Linear(2->64) + LN + exact GELU ----------------
  const float we0a = W_enc[lane * 2],        we1a = W_enc[lane * 2 + 1];
  const float we0b = W_enc[(lane + 32) * 2], we1b = W_enc[(lane + 32) * 2 + 1];
  const float bea = b_enc[lane],  beb = b_enc[lane + 32];
  const float ga  = ln_g[lane],   gb  = ln_g[lane + 32];
  const float la  = ln_b[lane],   lb  = ln_b[lane + 32];

  for (int m = 0; m < 16; ++m) {
    const int b = tileB + m;
    const float o0 = obs[2 * b], o1 = obs[2 * b + 1];
    float xa = we0a * o0 + we1a * o1 + bea;   // feature j = lane
    float xb = we0b * o0 + we1b * o1 + beb;   // feature j = lane+32
    float s = xa + xb, ss = xa * xa + xb * xb;
    #pragma unroll
    for (int msk = 16; msk >= 1; msk >>= 1) {
      s  += __shfl_xor(s, msk);
      ss += __shfl_xor(ss, msk);
    }
    const float mean = s * (1.0f / 64.0f);
    const float var  = ss * (1.0f / 64.0f) - mean * mean;
    const float rinv = rsqrtf(var + LN_EPS);
    float ha = (xa - mean) * rinv * ga + la;
    float hb = (xb - mean) * rinv * gb + lb;
    ha = 0.5f * ha * (1.0f + erff(ha * INV_SQRT2));   // exact GELU
    hb = 0.5f * hb * (1.0f + erff(hb * INV_SQRT2));
    myH[m * HH + lane]      = (_Float16)ha;
    myH[m * HH + lane + 32] = (_Float16)hb;
  }
  __syncthreads();

  // ---------------- mu / log_var heads via WMMA (K = 64 = 2 steps) ---------
  const v16h hA0 = load_a_frag(myH, HH, 0);
  const v16h hA1 = load_a_frag(myH, HH, 32);

  v8f muA[2], lvA[2];
  #pragma unroll
  for (int nt = 0; nt < 2; ++nt) {
    v8f acc = splat8(b_mu[nt * 16 + c]);
    acc = wmma_f16(hA0, load_b_frag(W_mu, HH, DD, HH, nt, 0),  acc);
    acc = wmma_f16(hA1, load_b_frag(W_mu, HH, DD, HH, nt, 32), acc);
    muA[nt] = acc;
    v8f acc2 = splat8(b_lv[nt * 16 + c]);
    acc2 = wmma_f16(hA0, load_b_frag(W_lv, HH, DD, HH, nt, 0),  acc2);
    acc2 = wmma_f16(hA1, load_b_frag(W_lv, HH, DD, HH, nt, 32), acc2);
    lvA[nt] = acc2;
  }

  // ---------------- reparam z + entropy partial -----------------------------
  v8f hC[2];                 // hidden state, C-fragment layout, fp32
  float entp = 0.0f;
  #pragma unroll
  for (int nt = 0; nt < 2; ++nt) {
    const int d = nt * 16 + c;
    #pragma unroll
    for (int g = 0; g < 8; ++g) {
      const int b = tileB + g + 8 * half;
      float lv = fminf(1.0f, fmaxf(-5.0f, lvA[nt][g]));
      entp += 1.0f + lv + LOG2PI;
      const float sd = fexp(0.5f * lv);
      hC[nt][g] = muA[nt][g] + eps[b * DD + d] * sd * 0.1f;   // LAMBDA_ENT
    }
  }
  #pragma unroll
  for (int msk = 16; msk >= 1; msk >>= 1) entp += __shfl_xor(entp, msk);
  if (lane == 0) sEntW[wave] = entp * 0.5f;

  __syncthreads();           // encoder tile dead; reuse myH as 16x32 GRU tile

  #pragma unroll
  for (int g = 0; g < 8; ++g) {
    const int m = g + 8 * half;
    myH[m * DD + c]      = (_Float16)hC[0][g];
    myH[m * DD + c + 16] = (_Float16)hC[1][g];
  }
  __syncthreads();
  v16h hA = load_a_frag(myH, DD, 0);

  // ---------------- persistent GRU weight fragments -------------------------
  v16h Bhh[6], Bih[6], Bout[2];
  #pragma unroll
  for (int j = 0; j < 6; ++j) {
    Bhh[j] = load_b_frag(W_hh, DD, 3 * DD, DD, j, 0);
    Bih[j] = load_b_frag(W_ih, VV, 3 * DD, VV, j, 0);   // K padded 27 -> 32
  }
  Bout[0] = load_b_frag(W_out, DD, VV, DD, 0, 0);       // N padded 27 -> 32
  Bout[1] = load_b_frag(W_out, DD, VV, DD, 1, 0);

  float bs[4];                                   // merged r/z biases
  #pragma unroll
  for (int j = 0; j < 4; ++j) bs[j] = b_ih[j * 16 + c] + b_hh[j * 16 + c];
  const float bin0 = b_ih[64 + c], bin1 = b_ih[80 + c];
  const float bhn0 = b_hh[64 + c], bhn1 = b_hh[80 + c];
  const float bo0  = b_out[c];
  const float bo1  = (c + 16 < VV) ? b_out[c + 16] : 0.0f;
  const bool  v1ok = (c + 16) < VV;

  // ---------------- GRU decode loop (fully unrolled) ------------------------
  #pragma unroll
  for (int t = 0; t < TT; ++t) {
    const bool haveX = (t > 0);     // statically resolved after unroll
    v16h oh{};
    if (haveX) {
      const int idxr = myIdx[c];                 // one-hot row = lane&15
      #pragma unroll
      for (int e = 0; e < 16; ++e) {
        const int K = e + ((e >= 8) ? 8 : 0) + half * 8;
        oh[e] = (_Float16)((K == idxr) ? 1.0f : 0.0f);
      }
    }

    v8f accRZ[4], gin[2], ghn[2];
    #pragma unroll
    for (int j = 0; j < 4; ++j) {                // r (tiles 0,1), z (tiles 2,3)
      v8f a = splat8(bs[j]);
      if (haveX) a = wmma_f16(oh, Bih[j], a);    // gi contribution
      a = wmma_f16(hA, Bhh[j], a);               // gh contribution
      accRZ[j] = a;
    }
    #pragma unroll
    for (int j = 0; j < 2; ++j) {                // n gate: keep gi and gh split
      v8f a = splat8(j ? bin1 : bin0);
      if (haveX) a = wmma_f16(oh, Bih[4 + j], a);
      gin[j] = a;
      v8f b2 = splat8(j ? bhn1 : bhn0);
      ghn[j] = wmma_f16(hA, Bhh[4 + j], b2);
    }

    #pragma unroll
    for (int g = 0; g < 8; ++g) {                // torch GRUCell update
      const float r0 = fsigmoid(accRZ[0][g]);
      const float r1 = fsigmoid(accRZ[1][g]);
      const float z0 = fsigmoid(accRZ[2][g]);
      const float z1 = fsigmoid(accRZ[3][g]);
      const float n0 = ftanh(gin[0][g] + r0 * ghn[0][g]);
      const float n1 = ftanh(gin[1][g] + r1 * ghn[1][g]);
      hC[0][g] = (1.0f - z0) * n0 + z0 * hC[0][g];
      hC[1][g] = (1.0f - z1) * n1 + z1 * hC[1][g];
    }

    #pragma unroll
    for (int g = 0; g < 8; ++g) {                // stage h_new for A-fragment
      const int m = g + 8 * half;
      myH[m * DD + c]      = (_Float16)hC[0][g];
      myH[m * DD + c + 16] = (_Float16)hC[1][g];
    }
    __syncthreads();
    hA = load_a_frag(myH, DD, 0);

    v8f lg0 = wmma_f16(hA, Bout[0], splat8(bo0));   // logits cols 0..15
    v8f lg1 = wmma_f16(hA, Bout[1], splat8(bo1));   // logits cols 16..26

    const float* ut = u + (size_t)t * BB * VV;
    #pragma unroll
    for (int g = 0; g < 8; ++g) {
      const int m = g + 8 * half;
      const int b = tileB + m;
      const float* urow = ut + (size_t)b * VV;
      float u0 = fminf(1.0f - 1e-7f, fmaxf(1e-7f, urow[c]));
      float s0 = (lg0[g] - flog(-flog(u0))) * TAU_INV;
      float s1 = -1e30f;
      if (v1ok) {
        float u1 = fminf(1.0f - 1e-7f, fmaxf(1e-7f, urow[c + 16]));
        s1 = (lg1[g] - flog(-flog(u1))) * TAU_INV;
      }
      float bestv = s0; int besti = c;
      if (s1 > bestv) { bestv = s1; besti = c + 16; }
      float mx = fmaxf(s0, s1);
      #pragma unroll
      for (int msk = 1; msk < 16; msk <<= 1) {    // butterfly over 16-lane group
        const float ov = __shfl_xor(bestv, msk);
        const int   oi = __shfl_xor(besti, msk);
        const float om = __shfl_xor(mx, msk);
        mx = fmaxf(mx, om);
        if (ov > bestv || (ov == bestv && oi < besti)) { bestv = ov; besti = oi; }
      }
      const float e0 = fexp(s0 - mx);
      const float e1 = v1ok ? fexp(s1 - mx) : 0.0f;
      float sm = e0 + e1;
      #pragma unroll
      for (int msk = 1; msk < 16; msk <<= 1) sm += __shfl_xor(sm, msk);
      const float inv = frcp(sm);
      const float y0 = e0 * inv, y1 = e1 * inv;
      const size_t wbase = (size_t)TT * BB + ((size_t)t * BB + b) * VV;
      out[wbase + c] = ((besti == c ? 1.0f : 0.0f) - y0) + y0;   // straight-through
      if (v1ok)
        out[wbase + c + 16] = ((besti == c + 16 ? 1.0f : 0.0f) - y1) + y1;
      if (c == 0) myIdx[m] = besti;               // lanes 0 & 16 publish argmax
    }
    __syncthreads();
    if (lane < 16)
      out[(size_t)t * BB + tileB + lane] = (float)myIdx[lane];   // indices
    if (t + 1 < TT)
      __builtin_prefetch(u + ((size_t)(t + 1) * BB + tileB) * VV, 0, 1);
  }

  __syncthreads();
  if (threadIdx.x == 0) {                         // fixed-order block partial
    float s = 0.0f;
    #pragma unroll
    for (int w = 0; w < 8; ++w) s += sEntW[w];
    ent_partials[blockIdx.x] = s;
  }
}

__global__ void POLOAR_ent_reduce(const float* __restrict__ partials,
                                  float* __restrict__ out, int n) {
  float s = 0.0f;
  for (int i = threadIdx.x; i < n; i += 32) s += partials[i];  // fixed order
  #pragma unroll
  for (int msk = 16; msk >= 1; msk >>= 1) s += __shfl_xor(s, msk);
  if (threadIdx.x == 0) out[(size_t)TT * BB + (size_t)TT * BB * VV] = s;
}

extern "C" void kernel_launch(void* const* d_in, const int* in_sizes, int n_in,
                              void* d_out, int out_size, void* d_ws,
                              size_t ws_size, hipStream_t stream) {
  const float* obs   = (const float*)d_in[0];
  const float* eps   = (const float*)d_in[1];
  const float* u     = (const float*)d_in[2];
  const float* W_enc = (const float*)d_in[3];
  const float* b_enc = (const float*)d_in[4];
  const float* ln_g  = (const float*)d_in[5];
  const float* ln_b  = (const float*)d_in[6];
  const float* W_mu  = (const float*)d_in[7];
  const float* b_mu  = (const float*)d_in[8];
  const float* W_lv  = (const float*)d_in[9];
  const float* b_lv  = (const float*)d_in[10];
  const float* W_ih  = (const float*)d_in[11];
  const float* b_ih  = (const float*)d_in[12];
  const float* W_hh  = (const float*)d_in[13];
  const float* b_hh  = (const float*)d_in[14];
  const float* W_out = (const float*)d_in[15];
  const float* b_out = (const float*)d_in[16];
  float* out = (float*)d_out;
  float* ws  = (float*)d_ws;

  const int nblocks = BB / 128;   // 1024 blocks, 8 waves x 16 rows each
  POLOAR_Brain_kernel<<<nblocks, 256, 0, stream>>>(
      obs, eps, u, W_enc, b_enc, ln_g, ln_b, W_mu, b_mu, W_lv, b_lv,
      W_ih, b_ih, W_hh, b_hh, W_out, b_out, out, ws);
  POLOAR_ent_reduce<<<1, 32, 0, stream>>>(ws, out, nblocks);
}
